// Model1_54760833024415
// MI455X (gfx1250) — compile-verified, matrix operand order checked by reading, not run on hardware
//
#include <hip/hip_runtime.h>

// ---------------- types ----------------
typedef __attribute__((ext_vector_type(16))) __bf16 bf16x16;
typedef __attribute__((ext_vector_type(8)))  __bf16 bf16x8;
typedef __attribute__((ext_vector_type(8)))  float  f32x8;
typedef __attribute__((ext_vector_type(4)))  int    i32x4;

#define BDIM 128
#define TDIM 20
#define HDIM 1024
#define H4   4096
#define KC   32

#define WMMA_BF16(A, B, C) __builtin_amdgcn_wmma_f32_16x16x32_bf16( \
    false, (A), false, (B), (short)0, (C), false, false)

// ---------------- CDNA5 async global->LDS copy (ASYNCcnt path) ----------------
#if defined(__gfx1250__) && __has_builtin(__builtin_amdgcn_global_load_async_to_lds_b128)
#define USE_ASYNC_LDS 1
#else
#define USE_ASYNC_LDS 0
#endif

// copy 16 bytes (8 x bf16) from global to LDS, per lane
__device__ __forceinline__ void copy16_g2l(const __bf16* g, __bf16* l) {
#if USE_ASYNC_LDS
    __builtin_amdgcn_global_load_async_to_lds_b128(
        (i32x4*)(void*)g, (i32x4*)(void*)l, 0, 0);
#else
    *(bf16x8*)l = *(const bf16x8*)g;   // synchronous fallback
#endif
}

__device__ __forceinline__ void async_drain() {
#if USE_ASYNC_LDS
#if __has_builtin(__builtin_amdgcn_s_wait_asynccnt)
    __builtin_amdgcn_s_wait_asynccnt(0);
#else
    asm volatile("s_wait_asynccnt 0x0" ::: "memory");
#endif
#endif
}

// ---------------- weight transpose + fp32 -> bf16 ----------------
// dst[n*K + k] = src[(row_off + k)*N + n]
__global__ __launch_bounds__(256) void transpose_f2bf(const float* __restrict__ src,
                                                      __bf16* __restrict__ dst,
                                                      int K, int N, int row_off) {
    long long id = (long long)blockIdx.x * 256 + threadIdx.x;
    long long total = (long long)K * N;
    if (id >= total) return;
    int n = (int)(id / K);
    int k = (int)(id % K);
    dst[(long long)n * K + k] = (__bf16)src[(long long)(row_off + k) * N + n];
}

// ---------------- layer-1 Zx (D == 1, rank-1) ----------------
__global__ __launch_bounds__(256) void zx1_kernel(const float* __restrict__ x,
                                                  const float* __restrict__ w1row0,
                                                  float* __restrict__ Zx) {
    long long id = (long long)blockIdx.x * 256 + threadIdx.x;
    long long total = (long long)TDIM * BDIM * H4;
    if (id >= total) return;
    int n = (int)(id % H4);
    int b = (int)((id / H4) % BDIM);
    int t = (int)(id / ((long long)H4 * BDIM));
    Zx[id] = x[b * TDIM + t] * w1row0[n];
}

// ---------------- generic bf16 WMMA GEMM: C = A @ B (async double-buffered) ----------------
// A: [M][K] bf16 row-major, BT: [N][K] bf16 (B transposed), C: [M][N] f32
// Block tile 128x128, 256 threads = 8 waves; wave w owns col-tile w, 8 row-tiles.
__global__ __launch_bounds__(256) void gemm_bf16(const __bf16* __restrict__ A,
                                                 const __bf16* __restrict__ BT,
                                                 float* __restrict__ C,
                                                 int M, int N, int K) {
    __shared__ __align__(32) __bf16 sA[2][128 * KC];
    __shared__ __align__(32) __bf16 sB[2][128 * KC];
    const int w    = threadIdx.x >> 5;
    const int lane = threadIdx.x & 31;
    const int m0 = blockIdx.x * 128;
    const int n0 = blockIdx.y * 128;

    const int row  = threadIdx.x >> 1;
    const int half = threadIdx.x & 1;
    const int soff = row * KC + half * 16;
    const __bf16* gA = &A[(long long)(m0 + row) * K + half * 16];
    const __bf16* gB = &BT[(long long)(n0 + row) * K + half * 16];

    const int h2  = lane >> 4;
    const int l16 = lane & 15;

    f32x8 acc[8] = {};

    auto stage = [&](int k0, int buf) {
        copy16_g2l(&gA[k0],     &sA[buf][soff]);
        copy16_g2l(&gA[k0 + 8], &sA[buf][soff + 8]);
        copy16_g2l(&gB[k0],     &sB[buf][soff]);
        copy16_g2l(&gB[k0 + 8], &sB[buf][soff + 8]);
    };
    auto compute = [&](const __bf16* bufA, const __bf16* bufB) {
        bf16x16 bfrag = *(const bf16x16*)&bufB[(w * 16 + l16) * KC + h2 * 16];
#pragma unroll
        for (int rt = 0; rt < 8; ++rt) {
            bf16x16 afrag = *(const bf16x16*)&bufA[(rt * 16 + l16) * KC + h2 * 16];
            acc[rt] = WMMA_BF16(afrag, bfrag, acc[rt]);
        }
    };

    stage(0, 0);
    async_drain();
    __syncthreads();

    int cur = 0;
    for (int k0 = KC; k0 < K; k0 += KC) {
        __builtin_prefetch(&gA[k0 + KC], 0, 3);
        __builtin_prefetch(&gB[k0 + KC], 0, 3);
        stage(k0, cur ^ 1);          // async copy in flight during the WMMAs
        compute(sA[cur], sB[cur]);
        async_drain();
        __syncthreads();             // single barrier per K-chunk
        cur ^= 1;
    }
    compute(sA[cur], sB[cur]);

#pragma unroll
    for (int rt = 0; rt < 8; ++rt)
#pragma unroll
        for (int v = 0; v < 8; ++v)
            C[(long long)(m0 + rt * 16 + h2 * 8 + v) * N + n0 + w * 16 + l16] = acc[rt][v];
}

// ---------------- fused recurrent step (async double-buffered) ----------------
// Each block owns 16 hidden units: z = Zx_t + h_prev @ Wh + b for the 4 gate
// column groups (128x64 f32), then the phased-LSTM elementwise update.
// Grid: H/16 = 64 blocks, 256 threads (8 waves).
__global__ __launch_bounds__(256) void plstm_step(
    const __bf16* __restrict__ hprev_bf,  // [B][H]
    float* __restrict__ c_state,          // [B][H] in/out (fp32)
    float* __restrict__ h_state,          // [B][H] in/out (fp32)
    __bf16* __restrict__ hout_bf,         // [B][H]
    const __bf16* __restrict__ WhT,       // [H4][H] (transposed recurrent weights)
    const float* __restrict__ Zx_t,       // [B][H4]
    const float* __restrict__ bias,       // [H4]
    const float* __restrict__ tvec,       // [B][T]
    int t_idx,
    const float* __restrict__ tau,
    const float* __restrict__ svec,
    const float* __restrict__ rvec) {

    __shared__ __align__(32) __bf16 sA[2][128 * KC];   // h_prev K-chunks
    __shared__ __align__(32) __bf16 sB[2][64 * KC];    // WhT chunks (64 gate-cols)
    __shared__ float sZ[128 * 64];                     // gate pre-activations

    const int w    = threadIdx.x >> 5;
    const int lane = threadIdx.x & 31;
    const int hblk = blockIdx.x;

    // A staging: thread -> (row, half)
    const int rowA  = threadIdx.x >> 1;
    const int halfA = threadIdx.x & 1;
    const int soffA = rowA * KC + halfA * 16;
    const __bf16* gA = &hprev_bf[(long long)rowA * HDIM + halfA * 16];

    // B staging: thread -> (gate-col cc, k-subchunk)
    const int cc    = threadIdx.x >> 2;        // 0..63
    const int kk0   = (threadIdx.x & 3) * 8;   // 0,8,16,24
    const int gcol  = (cc >> 4) * HDIM + hblk * 16 + (cc & 15);
    const int soffB = cc * KC + kk0;
    const __bf16* gB = &WhT[(long long)gcol * HDIM + kk0];

    const int h2  = lane >> 4;
    const int l16 = lane & 15;

    f32x8 acc[4] = {};

    auto stage = [&](int k0, int buf) {
        copy16_g2l(&gA[k0],     &sA[buf][soffA]);
        copy16_g2l(&gA[k0 + 8], &sA[buf][soffA + 8]);
        copy16_g2l(&gB[k0],     &sB[buf][soffB]);
    };
    auto compute = [&](const __bf16* bufA, const __bf16* bufB) {
        bf16x16 afrag = *(const bf16x16*)&bufA[(w * 16 + l16) * KC + h2 * 16];
#pragma unroll
        for (int ct = 0; ct < 4; ++ct) {
            bf16x16 bfrag = *(const bf16x16*)&bufB[(ct * 16 + l16) * KC + h2 * 16];
            acc[ct] = WMMA_BF16(afrag, bfrag, acc[ct]);
        }
    };

    stage(0, 0);
    async_drain();
    __syncthreads();

    int cur = 0;
    for (int k0 = KC; k0 < HDIM; k0 += KC) {
        __builtin_prefetch(&gA[k0 + KC], 0, 3);
        __builtin_prefetch(&gB[k0 + KC], 0, 3);
        stage(k0, cur ^ 1);          // async copy in flight during the WMMAs
        compute(sA[cur], sB[cur]);
        async_drain();
        __syncthreads();             // single barrier per K-chunk
        cur ^= 1;
    }
    compute(sA[cur], sB[cur]);

    // spill accumulators to LDS: row = w*16 + h2*8 + v, col = ct*16 + l16
#pragma unroll
    for (int ct = 0; ct < 4; ++ct)
#pragma unroll
        for (int v = 0; v < 8; ++v)
            sZ[(w * 16 + h2 * 8 + v) * 64 + ct * 16 + l16] = acc[ct][v];
    __syncthreads();

    // elementwise phased-LSTM update for this block's 128x16 (b, h) pairs
    const float LEAK = 0.001f;
    for (int i = threadIdx.x; i < 128 * 16; i += 256) {
        int b  = i >> 4;
        int hh = i & 15;
        int hg = hblk * 16 + hh;
        long long zb = (long long)b * H4;
        float zi = sZ[b * 64 +  0 + hh] + Zx_t[zb + hg]            + bias[hg];
        float zf = sZ[b * 64 + 16 + hh] + Zx_t[zb + HDIM + hg]     + bias[HDIM + hg];
        float zj = sZ[b * 64 + 32 + hh] + Zx_t[zb + 2 * HDIM + hg] + bias[2 * HDIM + hg];
        float zo = sZ[b * 64 + 48 + hh] + Zx_t[zb + 3 * HDIM + hg] + bias[3 * HDIM + hg];

        float c  = c_state[b * HDIM + hg];
        float hp = h_state[b * HDIM + hg];

        float si = 1.f / (1.f + __expf(-zi));
        float sf = 1.f / (1.f + __expf(-zf));
        float so = 1.f / (1.f + __expf(-zo));
        float ccand = sf * c + si * tanhf(zj);
        float hcand = tanhf(ccand) * so;

        float tt = tvec[b * TDIM + t_idx];
        float ta = tau[hg], ss = svec[hg], ro = rvec[hg];
        float phi = fmodf(tt - ss, ta);
        if (phi < 0.f) phi += ta;
        phi /= ta;
        float ku = 2.f * phi / ro;
        float kg = (phi < 0.5f * ro) ? ku : ((phi < ro) ? (2.f - ku) : LEAK * phi);

        float cn = kg * ccand + (1.f - kg) * c;
        float hn = kg * hcand + (1.f - kg) * hp;

        c_state[b * HDIM + hg] = cn;
        h_state[b * HDIM + hg] = hn;
        hout_bf[b * HDIM + hg] = (__bf16)hn;
    }
}

// ---------------- dense head: out[b] = sum_n tanh(Z[b][n]+bd1[n])*Wd2[n] + bd2 ----------------
__global__ __launch_bounds__(256) void head_kernel(const float* __restrict__ Z,
                                                   const float* __restrict__ bd1,
                                                   const float* __restrict__ Wd2,
                                                   const float* __restrict__ bd2,
                                                   float* __restrict__ out) {
    __shared__ float red[256];
    int b = blockIdx.x;
    float s = 0.f;
    for (int n = threadIdx.x; n < HDIM; n += 256)
        s += tanhf(Z[b * HDIM + n] + bd1[n]) * Wd2[n];
    red[threadIdx.x] = s;
    __syncthreads();
    for (int st = 128; st > 0; st >>= 1) {
        if (threadIdx.x < st) red[threadIdx.x] += red[threadIdx.x + st];
        __syncthreads();
    }
    if (threadIdx.x == 0) out[b] = red[0] + bd2[0];
}

// ---------------- host orchestration ----------------
extern "C" void kernel_launch(void* const* d_in, const int* in_sizes, int n_in,
                              void* d_out, int out_size, void* d_ws, size_t ws_size,
                              hipStream_t stream) {
    (void)in_sizes; (void)n_in; (void)out_size; (void)ws_size;
    const float* x    = (const float*)d_in[0];
    const float* tv   = (const float*)d_in[1];
    const float* W1   = (const float*)d_in[2];
    const float* b1   = (const float*)d_in[3];
    const float* tau1 = (const float*)d_in[4];
    const float* s1   = (const float*)d_in[5];
    const float* r1   = (const float*)d_in[6];
    const float* W2   = (const float*)d_in[7];
    const float* b2   = (const float*)d_in[8];
    const float* tau2 = (const float*)d_in[9];
    const float* s2   = (const float*)d_in[10];
    const float* r2   = (const float*)d_in[11];
    const float* W3   = (const float*)d_in[12];
    const float* b3   = (const float*)d_in[13];
    const float* tau3 = (const float*)d_in[14];
    const float* s3   = (const float*)d_in[15];
    const float* r3   = (const float*)d_in[16];
    const float* Wd1  = (const float*)d_in[17];
    const float* bd1  = (const float*)d_in[18];
    const float* Wd2  = (const float*)d_in[19];
    const float* bd2  = (const float*)d_in[20];
    float* out = (float*)d_out;

    // ---- workspace carve-up ----
    const size_t SZ_WT   = (size_t)H4 * HDIM * sizeof(__bf16);      // 8 MB
    const size_t SZ_WD1T = (size_t)HDIM * HDIM * sizeof(__bf16);    // 2 MB
    const size_t SZ_ZX   = (size_t)TDIM * BDIM * H4 * sizeof(float);// 40 MB
    const size_t SZ_HALL = (size_t)TDIM * BDIM * HDIM * sizeof(__bf16);
    const size_t SZ_H0   = (size_t)BDIM * HDIM * sizeof(__bf16);
    const size_t SZ_ST   = (size_t)BDIM * HDIM * sizeof(float);

    uint8_t* p = (uint8_t*)d_ws;
    __bf16* WhT1 = (__bf16*)p; p += SZ_WT;
    __bf16* WxT2 = (__bf16*)p; p += SZ_WT;
    __bf16* WhT2 = (__bf16*)p; p += SZ_WT;
    __bf16* WxT3 = (__bf16*)p; p += SZ_WT;
    __bf16* WhT3 = (__bf16*)p; p += SZ_WT;
    __bf16* Wd1T = (__bf16*)p; p += SZ_WD1T;
    float*  Zx   = (float*)p;  p += SZ_ZX;
    __bf16* hA   = (__bf16*)p; p += SZ_HALL;
    __bf16* hB   = (__bf16*)p; p += SZ_HALL;
    __bf16* h0   = (__bf16*)p; p += SZ_H0;
    float*  cst  = (float*)p;  p += SZ_ST;
    float*  hst  = (float*)p;  p += SZ_ST;

    // ---- 1. weight conversion / transposition (bf16, [N][K]) ----
    {
        int nblk_big = (int)(((long long)H4 * HDIM + 255) / 256);
        hipLaunchKernelGGL(transpose_f2bf, dim3(nblk_big), dim3(256), 0, stream, W1, WhT1, HDIM, H4, 1);
        hipLaunchKernelGGL(transpose_f2bf, dim3(nblk_big), dim3(256), 0, stream, W2, WxT2, HDIM, H4, 0);
        hipLaunchKernelGGL(transpose_f2bf, dim3(nblk_big), dim3(256), 0, stream, W2, WhT2, HDIM, H4, HDIM);
        hipLaunchKernelGGL(transpose_f2bf, dim3(nblk_big), dim3(256), 0, stream, W3, WxT3, HDIM, H4, 0);
        hipLaunchKernelGGL(transpose_f2bf, dim3(nblk_big), dim3(256), 0, stream, W3, WhT3, HDIM, H4, HDIM);
        int nblk_d1 = (int)(((long long)HDIM * HDIM + 255) / 256);
        hipLaunchKernelGGL(transpose_f2bf, dim3(nblk_d1), dim3(256), 0, stream, Wd1, Wd1T, HDIM, HDIM, 0);
    }

    (void)hipMemsetAsync(h0, 0, SZ_H0, stream);

    const long long stepB = (long long)BDIM * HDIM;   // per-t h elements
    const long long stepZ = (long long)BDIM * H4;     // per-t Zx elements

    // ---- layer 1 ----
    {
        int nblk = (int)(((long long)TDIM * BDIM * H4 + 255) / 256);
        hipLaunchKernelGGL(zx1_kernel, dim3(nblk), dim3(256), 0, stream, x, W1, Zx);
        (void)hipMemsetAsync(cst, 0, SZ_ST, stream);
        (void)hipMemsetAsync(hst, 0, SZ_ST, stream);
        for (int t = 0; t < TDIM; ++t) {
            const __bf16* hp = (t == 0) ? h0 : (hA + (long long)(t - 1) * stepB);
            hipLaunchKernelGGL(plstm_step, dim3(HDIM / 16), dim3(256), 0, stream,
                               hp, cst, hst, hA + (long long)t * stepB,
                               WhT1, Zx + (long long)t * stepZ, b1, tv, t, tau1, s1, r1);
        }
    }
    // ---- layer 2 ----
    {
        hipLaunchKernelGGL(gemm_bf16, dim3(TDIM * BDIM / 128, H4 / 128), dim3(256), 0, stream,
                           hA, WxT2, Zx, TDIM * BDIM, H4, HDIM);
        (void)hipMemsetAsync(cst, 0, SZ_ST, stream);
        (void)hipMemsetAsync(hst, 0, SZ_ST, stream);
        for (int t = 0; t < TDIM; ++t) {
            const __bf16* hp = (t == 0) ? h0 : (hB + (long long)(t - 1) * stepB);
            hipLaunchKernelGGL(plstm_step, dim3(HDIM / 16), dim3(256), 0, stream,
                               hp, cst, hst, hB + (long long)t * stepB,
                               WhT2, Zx + (long long)t * stepZ, b2, tv, t, tau2, s2, r2);
        }
    }
    // ---- layer 3 ----
    {
        hipLaunchKernelGGL(gemm_bf16, dim3(TDIM * BDIM / 128, H4 / 128), dim3(256), 0, stream,
                           hB, WxT3, Zx, TDIM * BDIM, H4, HDIM);
        (void)hipMemsetAsync(cst, 0, SZ_ST, stream);
        (void)hipMemsetAsync(hst, 0, SZ_ST, stream);
        for (int t = 0; t < TDIM; ++t) {
            const __bf16* hp = (t == 0) ? h0 : (hA + (long long)(t - 1) * stepB);
            hipLaunchKernelGGL(plstm_step, dim3(HDIM / 16), dim3(256), 0, stream,
                               hp, cst, hst, hA + (long long)t * stepB,
                               WhT3, Zx + (long long)t * stepZ, b3, tv, t, tau3, s3, r3);
        }
    }
    // ---- dense head ----
    {
        hipLaunchKernelGGL(gemm_bf16, dim3(1, HDIM / 128), dim3(256), 0, stream,
                           hA + (long long)(TDIM - 1) * stepB, Wd1T, Zx, BDIM, HDIM, HDIM);
        hipLaunchKernelGGL(head_kernel, dim3(BDIM), dim3(256), 0, stream, Zx, bd1, Wd2, bd2, out);
    }
}
